// GatedDeltaNetAttention_71622874628682
// MI455X (gfx1250) — matrix operand order repaired and checked by reading, not compiled, not run on hardware
//
#include <hip/hip_runtime.h>
#include <math.h>
#include <stdint.h>

// ---------------------------------------------------------------------------
// GatedDeltaNet for MI455X (gfx1250, wave32).  B=4 T=4096 D=1024 H=16 DH=64.
// Pipeline:
//   cvt_hi_lo    : f32 -> bf16 hi/lo splits (x, Wqkv, Wout)
//   gemm_bf16x3  : qkv = x @ Wqkv^T  (split-bf16 3-term WMMA, double-buffered
//                  GLOBAL_LOAD_ASYNC_TO_LDS_B128 staging, ASYNCcnt waits)
//   gates_kernel : alpha/beta gates
//   normalize_qk : l2norm of q,k heads (f32, in place)
//   chunk_prep   : per (bh,chunk=64): M = KK^T (WMMA), triangular solve for
//                  U,W (f32, column-parallel), P=W^T K~, G=U^T K~ (WMMA)
//   state_pass   : per bh sequential over 64 chunks: S = gL*S - S*P + G
//                  (one 64^3 WMMA matmul / step, async PT/G prefetch 2 ahead)
//   out_pass     : per (bh,chunk): O = D(U - W S0^T) + diag(g) Q S0^T (WMMA)
//   gemm_bf16x3  : out = o @ Wout^T
// ---------------------------------------------------------------------------

typedef __attribute__((ext_vector_type(8)))  float  v8f;
typedef __attribute__((ext_vector_type(16))) __bf16 v16bf;
typedef __attribute__((ext_vector_type(8)))  __bf16 v8bf;

union BF16x16 { v16bf v; v8bf h[2]; };

#define DIM   1024
#define HEADS 16
#define DH    64
#define BATCH 4
#define TLEN  4096
#define MTOT  (BATCH * TLEN)   // 16384 rows
#define CL    64               // chunk length
#define NCH   (TLEN / CL)      // 64 chunks
#define NBH   (BATCH * HEADS)  // 64
#define LD72  72               // padded bf16 row stride for 64-wide tiles

__device__ __forceinline__ void bsplit(float v, __bf16& h, __bf16& l) {
  h = (__bf16)v;
  l = (__bf16)(v - (float)h);
}

// --- CDNA5 async global->LDS (ASYNCcnt) -----------------------------------
// Generic shared-aperture addresses carry the LDS byte offset in addr[31:0]
// (ISA 10.2), so truncation yields the VDST LDS address for async loads.
__device__ __forceinline__ uint32_t lds32(const void* p) {
  return (uint32_t)(uintptr_t)p;
}
__device__ __forceinline__ void async_b128(uint32_t ldsdst, const void* gsrc) {
  asm volatile("global_load_async_to_lds_b128 %0, %1, off" ::"v"(ldsdst),
               "v"(gsrc)
               : "memory");
}
#if defined(__has_builtin)
#if __has_builtin(__builtin_amdgcn_s_wait_asynccnt)
#define HAVE_WAIT_ASYNC_BUILTIN 1
#endif
#endif
__device__ __forceinline__ void wait_async_le8() {
#ifdef HAVE_WAIT_ASYNC_BUILTIN
  __builtin_amdgcn_s_wait_asynccnt(8);
#else
  asm volatile("s_wait_asynccnt 0x8" ::: "memory");
#endif
}
__device__ __forceinline__ void wait_async_0() {
#ifdef HAVE_WAIT_ASYNC_BUILTIN
  __builtin_amdgcn_s_wait_asynccnt(0);
#else
  asm volatile("s_wait_asynccnt 0x0" ::: "memory");
#endif
}

// A-frag: A[row][k] at Asrc[row*lda + k]   (16x32 bf16 tile)
__device__ __forceinline__ BF16x16 ldfragA(const __bf16* p, int lda, int row0,
                                           int k0, int lr, int lh) {
  BF16x16 f;
  const __bf16* r = p + (size_t)(row0 + lr) * lda + k0 + lh * 8;
  f.h[0] = *(const v8bf*)r;
  f.h[1] = *(const v8bf*)(r + 16);
  return f;
}
// B-frag: B[k][col] at Bsrc[col*ldb + k]   (32x16 bf16 tile)
__device__ __forceinline__ BF16x16 ldfragB(const __bf16* p, int ldb, int col0,
                                           int k0, int lr, int lh) {
  BF16x16 f;
  const __bf16* r = p + (size_t)(col0 + lr) * ldb + k0 + lh * 8;
  f.h[0] = *(const v8bf*)r;
  f.h[1] = *(const v8bf*)(r + 16);
  return f;
}

#define WMMA3(acc, ah, al, bh_, bl_)                                          \
  acc = __builtin_amdgcn_wmma_f32_16x16x32_bf16(false, (ah).v, false, (bh_).v,\
                                                (short)0, acc, false, false); \
  acc = __builtin_amdgcn_wmma_f32_16x16x32_bf16(false, (ah).v, false, (bl_).v,\
                                                (short)0, acc, false, false); \
  acc = __builtin_amdgcn_wmma_f32_16x16x32_bf16(false, (al).v, false, (bh_).v,\
                                                (short)0, acc, false, false);

// 64x64x64 matmul, 2 output tiles per wave (16 tiles over 8 waves),
// split-bf16 3-term accumulation. A: [64][lda] rows; B: [64][ldb] "cols".
__device__ __forceinline__ void mm64x2(
    v8f* acc0, v8f* acc1, const __bf16* Ah, const __bf16* Al, int lda,
    const __bf16* Bh, const __bf16* Bl, int ldb, int mt, int nt0, int lr,
    int lh) {
#pragma unroll
  for (int kc = 0; kc < 2; ++kc) {
    const int k0 = kc * 32;
    BF16x16 ah = ldfragA(Ah, lda, mt * 16, k0, lr, lh);
    BF16x16 al = ldfragA(Al, lda, mt * 16, k0, lr, lh);
    BF16x16 b0h = ldfragB(Bh, ldb, nt0 * 16, k0, lr, lh);
    BF16x16 b0l = ldfragB(Bl, ldb, nt0 * 16, k0, lr, lh);
    BF16x16 b1h = ldfragB(Bh, ldb, (nt0 + 1) * 16, k0, lr, lh);
    BF16x16 b1l = ldfragB(Bl, ldb, (nt0 + 1) * 16, k0, lr, lh);
    WMMA3(*acc0, ah, al, b0h, b0l);
    WMMA3(*acc1, ah, al, b1h, b1l);
  }
}

// ---------------------------------------------------------------------------
// 1) f32 -> bf16 hi/lo
// ---------------------------------------------------------------------------
__global__ __launch_bounds__(256)
void cvt_hi_lo(const float* __restrict__ src, __bf16* __restrict__ hi,
               __bf16* __restrict__ lo, int n) {
  int i = blockIdx.x * 256 + threadIdx.x;
  if (i < n) {
    __bf16 h, l;
    bsplit(src[i], h, l);
    hi[i] = h;
    lo[i] = l;
  }
}

// ---------------------------------------------------------------------------
// 2) Split-bf16 GEMM: C(MxN,f32) = A(MxK) @ W(NxK)^T, K mult of 32, K/32 >= 2.
//    Double-buffered async global->LDS staging: 8 async b128 per thread per
//    K-chunk, two chunks in flight, s_wait_asynccnt 8/0 retires them in order.
// ---------------------------------------------------------------------------
#define LDK 40

__global__ __launch_bounds__(256)
void gemm_bf16x3(const __bf16* __restrict__ Ahg, const __bf16* __restrict__ Alg,
                 const __bf16* __restrict__ Bhg, const __bf16* __restrict__ Blg,
                 float* __restrict__ C, int M, int N, int K) {
  __shared__ __bf16 sAh[2][128 * LDK], sAl[2][128 * LDK];
  __shared__ __bf16 sBh[2][128 * LDK], sBl[2][128 * LDK];

  const int tid = threadIdx.x;
  const int bm = blockIdx.y * 128;
  const int bn = blockIdx.x * 128;
  const int lane = tid & 31;
  const int w = tid >> 5;
  const int lr = lane & 15;
  const int lh = lane >> 4;
  const int wm = (w & 3) * 32;
  const int wn = (w >> 2) * 64;

  v8f acc[2][4];
  const v8f vz = {0.f, 0.f, 0.f, 0.f, 0.f, 0.f, 0.f, 0.f};
#pragma unroll
  for (int i = 0; i < 2; ++i)
#pragma unroll
    for (int j = 0; j < 4; ++j) acc[i][j] = vz;

  const int ldr = tid >> 1;             // 0..127 row to stage
  const int ldc = (tid & 1) * 16;       // 0 / 16 k-offset
  const int off = ldr * LDK + ldc;

  uint32_t dAh[2] = {lds32(&sAh[0][off]), lds32(&sAh[1][off])};
  uint32_t dAl[2] = {lds32(&sAl[0][off]), lds32(&sAl[1][off])};
  uint32_t dBh[2] = {lds32(&sBh[0][off]), lds32(&sBh[1][off])};
  uint32_t dBl[2] = {lds32(&sBl[0][off]), lds32(&sBl[1][off])};

  auto issue = [&](int k0, int buf) {
    const size_t ga = (size_t)(bm + ldr) * K + k0 + ldc;
    const size_t gb = (size_t)(bn + ldr) * K + k0 + ldc;
    async_b128(dAh[buf], Ahg + ga);
    async_b128(dAh[buf] + 16, Ahg + ga + 8);
    async_b128(dAl[buf], Alg + ga);
    async_b128(dAl[buf] + 16, Alg + ga + 8);
    async_b128(dBh[buf], Bhg + gb);
    async_b128(dBh[buf] + 16, Bhg + gb + 8);
    async_b128(dBl[buf], Blg + gb);
    async_b128(dBl[buf] + 16, Blg + gb + 8);
  };

  const int nk = K / 32;
  issue(0, 0);
  issue(32, 1);

  for (int kc = 0; kc < nk; ++kc) {
    const int cur = kc & 1;
    if (kc + 1 < nk)
      wait_async_le8();                 // current tile's 8 loads retired
    else
      wait_async_0();
    __syncthreads();

    BF16x16 afh[2], afl[2];
#pragma unroll
    for (int mt = 0; mt < 2; ++mt) {
      afh[mt] = ldfragA(sAh[cur], LDK, wm + mt * 16, 0, lr, lh);
      afl[mt] = ldfragA(sAl[cur], LDK, wm + mt * 16, 0, lr, lh);
    }
#pragma unroll
    for (int nt = 0; nt < 4; ++nt) {
      BF16x16 bfh = ldfragB(sBh[cur], LDK, wn + nt * 16, 0, lr, lh);
      BF16x16 bfl = ldfragB(sBl[cur], LDK, wn + nt * 16, 0, lr, lh);
#pragma unroll
      for (int mt = 0; mt < 2; ++mt) {
        WMMA3(acc[mt][nt], afh[mt], afl[mt], bfh, bfl);
      }
    }
    __syncthreads();
    if (kc + 2 < nk) issue((kc + 2) * 32, cur);
  }

#pragma unroll
  for (int mt = 0; mt < 2; ++mt)
#pragma unroll
    for (int nt = 0; nt < 4; ++nt) {
      const int col = bn + wn + nt * 16 + lr;
#pragma unroll
      for (int rr = 0; rr < 8; ++rr) {
        const int row = bm + wm + mt * 16 + lh * 8 + rr;
        C[(size_t)row * N + col] = acc[mt][nt][rr];
      }
    }
}

// ---------------------------------------------------------------------------
// 3) gates
// ---------------------------------------------------------------------------
__global__ __launch_bounds__(256)
void gates_kernel(const float* __restrict__ x, const float* __restrict__ Wg,
                  const float* __restrict__ bg, const float* __restrict__ Wb,
                  const float* __restrict__ bb, float* __restrict__ gates) {
  __shared__ float xs[16 * DIM];
  const int m0 = blockIdx.x * 16;
  for (int idx = threadIdx.x; idx < 16 * DIM; idx += 256)
    xs[idx] = x[(size_t)m0 * DIM + idx];
  __syncthreads();

  const int r = threadIdx.x >> 4;
  const int g = threadIdx.x & 15;
  const float4* xr = (const float4*)&xs[r * DIM];
  const float4* wa = (const float4*)&Wg[(size_t)g * DIM];
  const float4* wb = (const float4*)&Wb[(size_t)g * DIM];
  float sa = 0.f, sb = 0.f;
#pragma unroll 4
  for (int k = 0; k < DIM / 4; ++k) {
    const float4 xv = xr[k];
    const float4 a4 = wa[k];
    const float4 b4 = wb[k];
    sa += xv.x * a4.x + xv.y * a4.y + xv.z * a4.z + xv.w * a4.w;
    sb += xv.x * b4.x + xv.y * b4.y + xv.z * b4.z + xv.w * b4.w;
  }
  const float alpha = 1.f / (1.f + __expf(-(sa + bg[g])));
  const float beta = 1.f / (1.f + __expf(-(sb + bb[g])));
  gates[(size_t)(m0 + r) * 32 + g] = alpha;
  gates[(size_t)(m0 + r) * 32 + 16 + g] = beta;
}

// ---------------------------------------------------------------------------
// 4) l2norm of q,k
// ---------------------------------------------------------------------------
__global__ __launch_bounds__(256)
void normalize_qk(float* __restrict__ qkv) {
  const int vid = blockIdx.x * 8 + (threadIdx.x >> 5);
  const int lane = threadIdx.x & 31;
  const int hh = vid & 15;
  const int p = (vid >> 4) & 1;
  const int m = vid >> 5;
  float* ptr = qkv + (size_t)m * (3 * DIM) + p * DIM + hh * DH + lane * 2;
  float2 e = *(float2*)ptr;
  float ss = e.x * e.x + e.y * e.y;
  ss += __shfl_xor(ss, 1);
  ss += __shfl_xor(ss, 2);
  ss += __shfl_xor(ss, 4);
  ss += __shfl_xor(ss, 8);
  ss += __shfl_xor(ss, 16);
  const float inv = 1.f / fmaxf(sqrtf(ss), 1e-12f);
  e.x *= inv;
  e.y *= inv;
  *(float2*)ptr = e;
}

// ---------------------------------------------------------------------------
// 5) chunk_prep (C1): one block per (bh,chunk).
// ---------------------------------------------------------------------------
__global__ __launch_bounds__(256)
void chunk_prep(const float* __restrict__ qkv, const float* __restrict__ gates,
                __bf16* __restrict__ UhG, __bf16* __restrict__ UlG,
                __bf16* __restrict__ WhG, __bf16* __restrict__ WlG,
                __bf16* __restrict__ PThG, __bf16* __restrict__ PTlG,
                __bf16* __restrict__ GhG, __bf16* __restrict__ GlG,
                float* __restrict__ gam) {
  __shared__ float La[CL], Gg[CL], Ge[CL], Bv[CL];
  __shared__ __bf16 sKh[CL * LD72], sKl[CL * LD72];
  __shared__ float Xs[CL * 128];                    // RHS / solution [t][0:128]
  __shared__ __bf16 sWTh[CL * LD72], sWTl[CL * LD72];
  __shared__ __bf16 sUTh[CL * LD72], sUTl[CL * LD72];
  // overlay: Mf (16KB, dead after solve) shares space with K~^T (18KB)
  __shared__ char ovl[2 * CL * LD72 * 2] __attribute__((aligned(16)));
  float* Mf = (float*)ovl;                          // [64][64]
  __bf16* sKtTh = (__bf16*)ovl;                     // [64][LD72]
  __bf16* sKtTl = (__bf16*)(ovl + CL * LD72 * 2);

  const int tid = threadIdx.x;
  const int bh = blockIdx.x >> 6;
  const int c = blockIdx.x & 63;
  const int b = bh >> 4, h = bh & 15;
  const int mbase = b * TLEN + c * CL;
  const size_t hcol = (size_t)h * DH;
  const int lane = tid & 31, wv = tid >> 5;
  const int lr = lane & 15, lh = lane >> 4;
  const int mt = wv >> 1, nt0 = (wv & 1) * 2;
  const v8f vz = {0.f, 0.f, 0.f, 0.f, 0.f, 0.f, 0.f, 0.f};

  if (tid < CL) {
    La[tid] = logf(gates[(size_t)(mbase + tid) * 32 + h]);
    Bv[tid] = gates[(size_t)(mbase + tid) * 32 + 16 + h];
  }
#pragma unroll
  for (int u = 0; u < 16; ++u) {                    // stage k (bf16 split)
    const int idx = tid + u * 256;
    const int t = idx >> 6, d = idx & 63;
    __bf16 hh, ll;
    bsplit(qkv[(size_t)(mbase + t) * (3 * DIM) + DIM + hcol + d], hh, ll);
    sKh[t * LD72 + d] = hh;
    sKl[t * LD72 + d] = ll;
  }
  __syncthreads();
  if (tid == 0) {                                   // cumulative log-decay
    float acc = 0.f;
    for (int t = 0; t < CL; ++t) {
      acc += La[t];
      Gg[t] = acc;
      Ge[t] = expf(acc);
    }
  }
  __syncthreads();

  // KK^T (WMMA) while staging RHS
  v8f a0 = vz, a1 = vz;
  mm64x2(&a0, &a1, sKh, sKl, LD72, sKh, sKl, LD72, mt, nt0, lr, lh);
#pragma unroll
  for (int u = 0; u < 32; ++u) {                    // X=[bV | b*gamma*K]
    const int idx = tid + u * 256;
    const int t = idx >> 7, cc = idx & 127;
    float val;
    if (cc < 64)
      val = Bv[t] * qkv[(size_t)(mbase + t) * (3 * DIM) + 2 * DIM + hcol + cc];
    else
      val = Bv[t] * Ge[t] *
            qkv[(size_t)(mbase + t) * (3 * DIM) + DIM + hcol + (cc - 64)];
    Xs[t * 128 + cc] = val;
  }
  {                                                  // M~ = b_t e^{g_t-g_s} KK^T
#pragma unroll
    for (int q2 = 0; q2 < 2; ++q2) {
      const v8f acc = q2 ? a1 : a0;
      const int s = (nt0 + q2) * 16 + lr;
#pragma unroll
      for (int e = 0; e < 8; ++e) {
        const int t = mt * 16 + lh * 8 + e;
        Mf[t * 64 + s] = (s < t) ? Bv[t] * expf(Gg[t] - Gg[s]) * acc[e] : 0.f;
      }
    }
  }
  __syncthreads();

  // forward substitution: columns independent -> no cross-thread deps
  if (tid < 128) {
    for (int t = 1; t < CL; ++t) {
      float acc = Xs[t * 128 + tid];
      for (int s = 0; s < t; ++s)
        acc = fmaf(-Mf[t * 64 + s], Xs[s * 128 + tid], acc);
      Xs[t * 128 + tid] = acc;
    }
  }
  __syncthreads();

  // export U,W (global bf16 hi/lo) + transposed LDS copies; stage K~^T
  const size_t ub = (size_t)blockIdx.x * (CL * DH);
#pragma unroll
  for (int u = 0; u < 16; ++u) {
    const int idx = tid + u * 256;
    const int t = idx >> 6, i = idx & 63;
    __bf16 hh, ll;
    bsplit(Xs[t * 128 + i], hh, ll);                 // U
    UhG[ub + idx] = hh;
    UlG[ub + idx] = ll;
    sUTh[i * LD72 + t] = hh;
    sUTl[i * LD72 + t] = ll;
    bsplit(Xs[t * 128 + 64 + i], hh, ll);            // W
    WhG[ub + idx] = hh;
    WlG[ub + idx] = ll;
    sWTh[i * LD72 + t] = hh;
    sWTl[i * LD72 + t] = ll;
    // K~^T[d][t] = e^{gL - g_t} k_t[d]   (overlays Mf; Mf dead)
    const float kt = expf(Gg[63] - Gg[t]) *
                     qkv[(size_t)(mbase + t) * (3 * DIM) + DIM + hcol + i];
    bsplit(kt, hh, ll);
    sKtTh[i * LD72 + t] = hh;
    sKtTl[i * LD72 + t] = ll;
  }
  __syncthreads();

  // P = W^T K~  (store transposed: PT[j][kk]);  G = U^T K~ (row-major)
  v8f p0 = vz, p1 = vz, g0 = vz, g1 = vz;
  mm64x2(&p0, &p1, sWTh, sWTl, LD72, sKtTh, sKtTl, LD72, mt, nt0, lr, lh);
  mm64x2(&g0, &g1, sUTh, sUTl, LD72, sKtTh, sKtTl, LD72, mt, nt0, lr, lh);
#pragma unroll
  for (int q2 = 0; q2 < 2; ++q2) {
    const v8f pp = q2 ? p1 : p0;
    const v8f gg = q2 ? g1 : g0;
    const int j = (nt0 + q2) * 16 + lr;
    const int r0 = mt * 16 + lh * 8;
    v8bf ph, pl;
#pragma unroll
    for (int e = 0; e < 8; ++e) {
      __bf16 hh, ll;
      bsplit(pp[e], hh, ll);
      ph[e] = hh;
      pl[e] = ll;
      bsplit(gg[e], hh, ll);
      GhG[ub + (size_t)(r0 + e) * 64 + j] = hh;
      GlG[ub + (size_t)(r0 + e) * 64 + j] = ll;
    }
    *(v8bf*)(PThG + ub + (size_t)j * 64 + r0) = ph;
    *(v8bf*)(PTlG + ub + (size_t)j * 64 + r0) = pl;
  }
  if (tid == 0) gam[blockIdx.x] = Ge[63];
}

// ---------------------------------------------------------------------------
// 6) state_pass (C2): one block per bh, sequential over 64 chunks.
//    S_next = gamma_L*S - S*P + G.  PT/G staged by async global->LDS copies,
//    double-buffered, issued two chunks ahead.
// ---------------------------------------------------------------------------
__global__ __launch_bounds__(256)
void state_pass(const __bf16* __restrict__ PThG, const __bf16* __restrict__ PTlG,
                const __bf16* __restrict__ GhG, const __bf16* __restrict__ GlG,
                const float* __restrict__ gam, __bf16* __restrict__ S0hG,
                __bf16* __restrict__ S0lG) {
  __shared__ float Sf[CL * 64];
  __shared__ __bf16 sSh[CL * LD72], sSl[CL * LD72];
  __shared__ __bf16 sPTh[2][CL * LD72], sPTl[2][CL * LD72];
  __shared__ __bf16 sGh[2][CL * 64], sGl[2][CL * 64];

  const int tid = threadIdx.x;
  const int bh = blockIdx.x;
  const int lane = tid & 31, wv = tid >> 5;
  const int lr = lane & 15, lh = lane >> 4;
  const int mt = wv >> 1, nt0 = (wv & 1) * 2;
  const v8f vz = {0.f, 0.f, 0.f, 0.f, 0.f, 0.f, 0.f, 0.f};

  const int e0 = tid * 16;                 // 16 contiguous elems per thread
  const int lo72 = (e0 >> 6) * LD72 + (e0 & 63);

#pragma unroll
  for (int u = 0; u < 16; ++u) {
    const int idx = tid + u * 256;
    Sf[idx] = 0.f;
    sSh[(idx >> 6) * LD72 + (idx & 63)] = (__bf16)0.f;
    sSl[(idx >> 6) * LD72 + (idx & 63)] = (__bf16)0.f;
  }

  uint32_t dPh[2] = {lds32(&sPTh[0][lo72]), lds32(&sPTh[1][lo72])};
  uint32_t dPl[2] = {lds32(&sPTl[0][lo72]), lds32(&sPTl[1][lo72])};
  uint32_t dGh[2] = {lds32(&sGh[0][e0]), lds32(&sGh[1][e0])};
  uint32_t dGl[2] = {lds32(&sGl[0][e0]), lds32(&sGl[1][e0])};

  auto issue = [&](int c, int buf) {
    const size_t base = ((size_t)bh * NCH + c) * (CL * DH) + e0;
    async_b128(dPh[buf], PThG + base);
    async_b128(dPh[buf] + 16, PThG + base + 8);
    async_b128(dPl[buf], PTlG + base);
    async_b128(dPl[buf] + 16, PTlG + base + 8);
    async_b128(dGh[buf], GhG + base);
    async_b128(dGh[buf] + 16, GhG + base + 8);
    async_b128(dGl[buf], GlG + base);
    async_b128(dGl[buf] + 16, GlG + base + 8);
  };

  issue(0, 0);
  issue(1, 1);

  for (int c = 0; c < NCH; ++c) {
    const int cur = c & 1;
    if (c + 1 < NCH)
      wait_async_le8();
    else
      wait_async_0();
    __syncthreads();

    {  // S0[c] = state before this chunk (copy LDS hi/lo -> global)
      const size_t base = ((size_t)bh * NCH + c) * (CL * DH) + e0;
#pragma unroll
      for (int j = 0; j < 2; ++j) {
        *(v8bf*)(S0hG + base + j * 8) = *(const v8bf*)&sSh[lo72 + j * 8];
        *(v8bf*)(S0lG + base + j * 8) = *(const v8bf*)&sSl[lo72 + j * 8];
      }
    }
    v8f a0 = vz, a1 = vz;                  // S * P
    mm64x2(&a0, &a1, sSh, sSl, LD72, sPTh[cur], sPTl[cur], LD72, mt, nt0, lr,
           lh);
    const float gL = gam[bh * NCH + c];
    __syncthreads();                       // all reads of sSh / sPT[cur] done
#pragma unroll
    for (int q2 = 0; q2 < 2; ++q2) {       // S = gL*S - S*P + G
      const v8f acc = q2 ? a1 : a0;
      const int j = (nt0 + q2) * 16 + lr;
#pragma unroll
      for (int e = 0; e < 8; ++e) {
        const int i = mt * 16 + lh * 8 + e;
        const int idx = i * 64 + j;
        const float gv = (float)sGh[cur][idx] + (float)sGl[cur][idx];
        const float ns = gL * Sf[idx] - acc[e] + gv;
        Sf[idx] = ns;
        __bf16 hh, ll;
        bsplit(ns, hh, ll);
        sSh[i * LD72 + j] = hh;
        sSl[i * LD72 + j] = ll;
      }
    }
    if (c + 2 < NCH) issue(c + 2, cur);    // buf cur free: reads done above
    __syncthreads();
  }
}

// ---------------------------------------------------------------------------
// 7) out_pass (C3): one block per (bh,chunk).
//    O = D*(U - W S0^T) + diag(gamma_t)*(Q S0^T); D[t][s]=e^{g_t-g_s}(q.k), s<=t
// ---------------------------------------------------------------------------
__global__ __launch_bounds__(256)
void out_pass(const float* __restrict__ qkv, const float* __restrict__ gates,
              const __bf16* __restrict__ UhG, const __bf16* __restrict__ UlG,
              const __bf16* __restrict__ WhG, const __bf16* __restrict__ WlG,
              const __bf16* __restrict__ S0hG, const __bf16* __restrict__ S0lG,
              __bf16* __restrict__ ohG, __bf16* __restrict__ olG) {
  __shared__ float La[CL], Gg[CL], Ge[CL];
  __shared__ __bf16 sQh[CL * LD72], sQl[CL * LD72];
  __shared__ __bf16 sDh[CL * LD72], sDl[CL * LD72];
  __shared__ __bf16 sS0h[CL * LD72], sS0l[CL * LD72];
  __shared__ float Uf[CL * 64];
  // overlay: K (dead after QK^T) shares space with E^T
  __shared__ char ovl[2 * CL * LD72 * 2] __attribute__((aligned(16)));
  __bf16* sKh = (__bf16*)ovl;
  __bf16* sKl = (__bf16*)(ovl + CL * LD72 * 2);
  __bf16* sETh = sKh;
  __bf16* sETl = sKl;

  const int tid = threadIdx.x;
  const int bh = blockIdx.x >> 6;
  const int c = blockIdx.x & 63;
  const int b = bh >> 4, h = bh & 15;
  const int mbase = b * TLEN + c * CL;
  const size_t hcol = (size_t)h * DH;
  const size_t ub = (size_t)blockIdx.x * (CL * DH);
  const int lane = tid & 31, wv = tid >> 5;
  const int lr = lane & 15, lh = lane >> 4;
  const int mt = wv >> 1, nt0 = (wv & 1) * 2;
  const v8f vz = {0.f, 0.f, 0.f, 0.f, 0.f, 0.f, 0.f, 0.f};

  if (tid < CL) La[tid] = logf(gates[(size_t)(mbase + tid) * 32 + h]);
  {
    const int e0 = tid * 16;
    const int lo72 = (e0 >> 6) * LD72 + (e0 & 63);
#pragma unroll
    for (int j = 0; j < 2; ++j) {
      *(v8bf*)&sS0h[lo72 + j * 8] = *(const v8bf*)(S0hG + ub + e0 + j * 8);
      *(v8bf*)&sS0l[lo72 + j * 8] = *(const v8bf*)(S0lG + ub + e0 + j * 8);
    }
  }
#pragma unroll
  for (int u = 0; u < 16; ++u) {
    const int idx = tid + u * 256;
    const int t = idx >> 6, d = idx & 63;
    __bf16 hh, ll;
    bsplit(qkv[(size_t)(mbase + t) * (3 * DIM) + hcol + d], hh, ll);
    sQh[t * LD72 + d] = hh;
    sQl[t * LD72 + d] = ll;
    bsplit(qkv[(size_t)(mbase + t) * (3 * DIM) + DIM + hcol + d], hh, ll);
    sKh[t * LD72 + d] = hh;
    sKl[t * LD72 + d] = ll;
    Uf[idx] = (float)UhG[ub + idx] + (float)UlG[ub + idx];
  }
  __syncthreads();
  if (tid == 0) {
    float acc = 0.f;
    for (int t = 0; t < CL; ++t) {
      acc += La[t];
      Gg[t] = acc;
      Ge[t] = expf(acc);
    }
  }
  __syncthreads();

  v8f qa0 = vz, qa1 = vz;                  // Q K^T
  mm64x2(&qa0, &qa1, sQh, sQl, LD72, sKh, sKl, LD72, mt, nt0, lr, lh);
  v8f fa0 = vz, fa1 = vz;                  // F = W S0^T (A direct from global)
  mm64x2(&fa0, &fa1, WhG + ub, WlG + ub, 64, sS0h, sS0l, LD72, mt, nt0, lr, lh);
  __syncthreads();                         // K-frag reads done (ET overlays K)

#pragma unroll
  for (int q2 = 0; q2 < 2; ++q2) {
    const v8f qq = q2 ? qa1 : qa0;
    const v8f ff = q2 ? fa1 : fa0;
    const int col = (nt0 + q2) * 16 + lr;
    const int r0 = mt * 16 + lh * 8;
#pragma unroll
    for (int e = 0; e < 8; ++e) {          // D masked + decayed
      const int t = r0 + e;
      const float dv = (col <= t) ? expf(Gg[t] - Gg[col]) * qq[e] : 0.f;
      __bf16 hh, ll;
      bsplit(dv, hh, ll);
      sDh[t * LD72 + col] = hh;
      sDl[t * LD72 + col] = ll;
    }
    v8bf eh, el;                           // E^T[i][t] = U[t][i] - F[t][i]
#pragma unroll
    for (int e = 0; e < 8; ++e) {
      __bf16 hh, ll;
      bsplit(Uf[(r0 + e) * 64 + col] - ff[e], hh, ll);
      eh[e] = hh;
      el[e] = ll;
    }
    *(v8bf*)&sETh[col * LD72 + r0] = eh;
    *(v8bf*)&sETl[col * LD72 + r0] = el;
  }
  __syncthreads();

  v8f oa0 = vz, oa1 = vz;                  // D * E
  mm64x2(&oa0, &oa1, sDh, sDl, LD72, sETh, sETl, LD72, mt, nt0, lr, lh);
  v8f sa0 = vz, sa1 = vz;                  // Q * S0^T
  mm64x2(&sa0, &sa1, sQh, sQl, LD72, sS0h, sS0l, LD72, mt, nt0, lr, lh);

#pragma unroll
  for (int q2 = 0; q2 < 2; ++q2) {
    const v8f ao = q2 ? oa1 : oa0;
    const v8f as = q2 ? sa1 : sa0;
    const int col = (nt0 + q2) * 16 + lr;
#pragma unroll
    for (int e = 0; e < 8; ++e) {
      const int t = mt * 16 + lh * 8 + e;
      const float ov = ao[e] + Ge[t] * as[e];
      const size_t oi = (size_t)(mbase + t) * DIM + hcol + col;
      __bf16 hh, ll;
      bsplit(ov, hh, ll);
      ohG[oi] = hh;
      olG[oi] = ll;
    }
  }
}

// ---------------------------------------------------------------------------
// launcher
// ---------------------------------------------------------------------------
extern "C" void kernel_launch(void* const* d_in, const int* in_sizes, int n_in,
                              void* d_out, int out_size, void* d_ws,
                              size_t ws_size, hipStream_t stream) {
  const float* x = (const float*)d_in[0];
  const float* Wqkv = (const float*)d_in[1];
  const float* Wg = (const float*)d_in[2];
  const float* bg = (const float*)d_in[3];
  const float* Wb = (const float*)d_in[4];
  const float* bb = (const float*)d_in[5];
  const float* Wout = (const float*)d_in[6];
  float* out = (float*)d_out;

  const size_t CHE = (size_t)NBH * NCH * CL * DH;

  char* p = (char*)d_ws;
  float* qkv = (float*)p;      p += (size_t)MTOT * 3 * DIM * 4;
  float* gates = (float*)p;    p += (size_t)MTOT * 32 * 4;
  __bf16* xh = (__bf16*)p;     p += (size_t)MTOT * DIM * 2;
  __bf16* xl = (__bf16*)p;     p += (size_t)MTOT * DIM * 2;
  __bf16* wqh = (__bf16*)p;    p += (size_t)3 * DIM * DIM * 2;
  __bf16* wql = (__bf16*)p;    p += (size_t)3 * DIM * DIM * 2;
  __bf16* woh = (__bf16*)p;    p += (size_t)DIM * DIM * 2;
  __bf16* wol = (__bf16*)p;    p += (size_t)DIM * DIM * 2;
  __bf16* oh = (__bf16*)p;     p += (size_t)MTOT * DIM * 2;
  __bf16* ol = (__bf16*)p;     p += (size_t)MTOT * DIM * 2;
  __bf16* Uh = (__bf16*)p;     p += CHE * 2;
  __bf16* Ul = (__bf16*)p;     p += CHE * 2;
  __bf16* Wh = (__bf16*)p;     p += CHE * 2;
  __bf16* Wl = (__bf16*)p;     p += CHE * 2;
  __bf16* PTh = (__bf16*)p;    p += CHE * 2;
  __bf16* PTl = (__bf16*)p;    p += CHE * 2;
  __bf16* Gh = (__bf16*)p;     p += CHE * 2;
  __bf16* Gl = (__bf16*)p;     p += CHE * 2;
  __bf16* S0h = (__bf16*)p;    p += CHE * 2;
  __bf16* S0l = (__bf16*)p;    p += CHE * 2;
  float* gam = (float*)p;      p += (size_t)NBH * NCH * 4;

  cvt_hi_lo<<<(MTOT * DIM) / 256, 256, 0, stream>>>(x, xh, xl, MTOT * DIM);
  cvt_hi_lo<<<(3 * DIM * DIM) / 256, 256, 0, stream>>>(Wqkv, wqh, wql,
                                                       3 * DIM * DIM);
  cvt_hi_lo<<<(DIM * DIM) / 256, 256, 0, stream>>>(Wout, woh, wol, DIM * DIM);

  gemm_bf16x3<<<dim3(3 * DIM / 128, MTOT / 128), 256, 0, stream>>>(
      xh, xl, wqh, wql, qkv, MTOT, 3 * DIM, DIM);

  gates_kernel<<<MTOT / 16, 256, 0, stream>>>(x, Wg, bg, Wb, bb, gates);
  normalize_qk<<<(2 * MTOT * HEADS) / 8, 256, 0, stream>>>(qkv);

  chunk_prep<<<NBH * NCH, 256, 0, stream>>>(qkv, gates, Uh, Ul, Wh, Wl, PTh,
                                            PTl, Gh, Gl, gam);
  state_pass<<<NBH, 256, 0, stream>>>(PTh, PTl, Gh, Gl, gam, S0h, S0l);
  out_pass<<<NBH * NCH, 256, 0, stream>>>(qkv, gates, Uh, Ul, Wh, Wl, S0h, S0l,
                                          oh, ol);

  gemm_bf16x3<<<dim3(DIM / 128, MTOT / 128), 256, 0, stream>>>(
      oh, ol, woh, wol, out, MTOT, DIM, DIM);
}
